// Conv2dWeightModulate_45509473469072
// MI455X (gfx1250) — matrix operand order repaired and checked by reading, not compiled, not run on hardware
//
#include <hip/hip_runtime.h>
#include <hip/hip_bf16.h>

// ---- problem constants (from reference) ----
#define IN_F   512
#define OUT_F  512
#define HW     32
#define PIX    (HW*HW)          // 1024
#define KTAPS  9
#define RDIM   (IN_F*KTAPS)     // 4608
#define KSTEPS (RDIM/32)        // 144
#define RUNTIME_COEF 0.014731391274719742f   // 1/sqrt(512*9)
#define EPSF   1e-8f

typedef __attribute__((ext_vector_type(16))) __bf16 v16bf;
typedef __attribute__((ext_vector_type(8)))  float  v8f;

// ------------------------------------------------------------------
// Kernel 1: per-(b,o) demodulation statistic, fp32-exact.
//   g[b,o] = coef * rsqrt(coef^2 * sum_i,k (weight[o,i,k]*style[b,i])^2 + eps)
// Final GEMM A element = weight * style * g  (then cast to bf16).
// ------------------------------------------------------------------
__global__ __launch_bounds__(256)
void modstats_kernel(const float* __restrict__ style,
                     const float* __restrict__ weight,
                     float* __restrict__ g)
{
    __shared__ float red[256];
    const int bo = blockIdx.x;          // 0..4095
    const int b  = bo >> 9;
    const int o  = bo & 511;
    const float* wrow = weight + (size_t)o * RDIM;   // [i*9 + tap]
    const float* st   = style + b * IN_F;

    float sum = 0.0f;
    for (int idx = threadIdx.x; idx < RDIM; idx += 256) {
        const int i = idx / 9;
        const float v = wrow[idx] * st[i];
        sum += v * v;
    }
    red[threadIdx.x] = sum;
    __syncthreads();
    for (int s = 128; s > 0; s >>= 1) {
        if (threadIdx.x < s) red[threadIdx.x] += red[threadIdx.x + s];
        __syncthreads();
    }
    if (threadIdx.x == 0) {
        const float c2 = RUNTIME_COEF * RUNTIME_COEF;
        g[bo] = RUNTIME_COEF * rsqrtf(red[0] * c2 + EPSF);
    }
}

// ------------------------------------------------------------------
// Kernel 2: implicit GEMM, out[b][o][p] = sum_r Wm[b][o][r] * col[b][r][p]
//   r = tap*512 + ic ; block tile M=128 (o) x N=64 (pixels), K-step = 32.
//   8 wave32s arranged 4(M) x 2(N); each wave: 2x2 tiles of 16x16 via
//   v_wmma_f32_16x16x32_bf16 with fp32 accumulation.
// LDS tiles are stored pre-swizzled per the CDNA5 fragment layouts so each
// lane's operand is one contiguous 32B read.
// ------------------------------------------------------------------
__global__ __launch_bounds__(256)
void modconv_gemm_kernel(const float* __restrict__ x,
                         const float* __restrict__ style,
                         const float* __restrict__ weight,
                         const float* __restrict__ g,
                         float* __restrict__ out)
{
    // A: 8 subtiles(16 rows) x 32 lanes x 16 bf16 = 128x32 tile, 8 KB
    // B: 4 subtiles(16 cols) x 32 lanes x 16 bf16 =  32x64 tile, 4 KB
    __shared__ alignas(32) __bf16 ldsA[8 * 32 * 16];
    __shared__ alignas(32) __bf16 ldsB[4 * 32 * 16];

    const int tid  = threadIdx.x;
    const int lane = tid & 31;
    const int wave = tid >> 5;
    const int wm   = wave & 3;   // 0..3  -> M quadrant (32 rows)
    const int wn   = wave >> 2;  // 0..1  -> N half     (32 cols)

    const int b     = blockIdx.z;
    const int oBase = blockIdx.y * 128;
    const int pBase = blockIdx.x * 64;

    // ---- A-build coordinates (all 256 threads; one fragment row each) ----
    // ISA 16-bit A 16x32 layout: lane<16: M=lane,  j<8 -> K=j,   j>=8 -> K=j+8
    //                            lane>=16: M=lane-16, add 8 to K.
    const int aM     = lane & 15;
    const int aKhalf = (lane >> 4) << 3;            // 0 or 8
    const int amb    = wave;                        // M subtile 0..7
    const int ao     = oBase + amb * 16 + aM;       // global output channel
    const float ag   = g[b * OUT_F + ao];           // invariant over K loop
    const float* wrow = weight + (size_t)ao * RDIM;
    const float* st   = style + b * IN_F;

    // ---- B-build coordinates (threads 0..127; one fragment row each) ----
    // Dense 16-bit B 32x16 layout: N=lane&15, K order same as A (j<8 -> K=j, j>=8 -> K=j+8, +8 if lane>=16)
    const int bnb   = tid >> 5;                     // N subtile 0..3 (tid<128)
    const int bN    = (lane & 15) + bnb * 16;       // 0..63 within block tile
    const int bKoff = (lane >> 4) << 4;             // 0 or 16
    const int bp    = pBase + bN;
    const int bpy   = bp >> 5;
    const int bpx   = bp & 31;
    const float* xb = x + (size_t)b * IN_F * PIX;

    v8f acc[2][2] = {};

    for (int ks = 0; ks < KSTEPS; ++ks) {
        const int tap = ks >> 4;               // 0..8
        const int ic0 = (ks & 15) << 5;        // channel chunk base, 0..480
        const int dy  = tap / 3 - 1;
        const int dx  = tap % 3 - 1;

        // -------- stage A tile (modulate + demodulate + cvt bf16) --------
        {
            __bf16* dst = &ldsA[(amb * 32 + lane) * 16];
            #pragma unroll
            for (int j = 0; j < 16; ++j) {
                const int Kv = ((j < 8) ? j : j + 8) + aKhalf;
                const int i  = ic0 + Kv;
                const float v = wrow[i * 9 + tap] * st[i] * ag;
                dst[j] = (__bf16)v;
            }
            // hint: prefetch next chunk of this weight row into cache
            if (ks + 1 < KSTEPS) {
                const int nic0 = ((ks + 1) & 15) << 5;
                __builtin_prefetch(&wrow[(nic0 + aKhalf) * 9 + ((ks + 1) >> 4)], 0, 1);
            }
        }

        // -------- stage B tile (im2col shifted read + cvt bf16) --------
        if (tid < 128) {
            __bf16* dst = &ldsB[(bnb * 32 + lane) * 16];
            const int ys = bpy + dy;
            const int xs = bpx + dx;
            const bool inb = ((unsigned)ys < 32u) && ((unsigned)xs < 32u);
            #pragma unroll
            for (int j = 0; j < 16; ++j) {
                const int ic = ic0 + ((j < 8) ? j : j + 8) + (bKoff >> 1);
                const float v = inb ? xb[(size_t)ic * PIX + ys * HW + xs] : 0.0f;
                dst[j] = (__bf16)v;
            }
        }
        __syncthreads();

        // -------- 2x2 WMMA tiles per wave --------
        v16bf afrag[2], bfrag[2];
        #pragma unroll
        for (int mi = 0; mi < 2; ++mi)
            afrag[mi] = *(const v16bf*)&ldsA[((wm * 2 + mi) * 32 + lane) * 16];
        #pragma unroll
        for (int ni = 0; ni < 2; ++ni)
            bfrag[ni] = *(const v16bf*)&ldsB[((wn * 2 + ni) * 32 + lane) * 16];

        #pragma unroll
        for (int mi = 0; mi < 2; ++mi)
            #pragma unroll
            for (int ni = 0; ni < 2; ++ni)
                acc[mi][ni] = __builtin_amdgcn_wmma_f32_16x16x32_bf16(
                    /*neg_a=*/false, afrag[mi],
                    /*neg_b=*/false, bfrag[ni],
                    /*c_mod=*/(short)0, acc[mi][ni],
                    /*reuse_a=*/false, /*reuse_b=*/false);

        __syncthreads();
    }

    // -------- epilogue: fp32 store per C/D layout --------
    // VGPR v, lane: row M = v + 8*(lane>=16), col N = lane&15
    const int hi = lane >> 4;
    const int nc = lane & 15;
    #pragma unroll
    for (int mi = 0; mi < 2; ++mi) {
        const int orow0 = oBase + (wm * 2 + mi) * 16 + hi * 8;
        #pragma unroll
        for (int ni = 0; ni < 2; ++ni) {
            const int pcol = pBase + (wn * 2 + ni) * 16 + nc;
            float* op = out + ((size_t)b * OUT_F + orow0) * PIX + pcol;
            #pragma unroll
            for (int v = 0; v < 8; ++v)
                op[(size_t)v * PIX] = acc[mi][ni][v];
        }
    }
}

// ------------------------------------------------------------------
extern "C" void kernel_launch(void* const* d_in, const int* in_sizes, int n_in,
                              void* d_out, int out_size, void* d_ws, size_t ws_size,
                              hipStream_t stream)
{
    const float* x      = (const float*)d_in[0];  // [8,512,32,32]
    const float* style  = (const float*)d_in[1];  // [8,512]
    const float* weight = (const float*)d_in[2];  // [512,512,3,3]
    float* out = (float*)d_out;                   // [8,512,32,32]
    float* g   = (float*)d_ws;                    // needs 8*512*4 = 16 KB

    (void)in_sizes; (void)n_in; (void)out_size; (void)ws_size;

    // Phase 1: fp32-exact demodulation scale per (b, out_channel)
    modstats_kernel<<<8 * OUT_F, 256, 0, stream>>>(style, weight, g);

    // Phase 2: implicit GEMM, bf16 WMMA with fp32 accumulation
    dim3 grid(PIX / 64, OUT_F / 128, 8);   // (16, 4, 8)
    modconv_gemm_kernel<<<grid, 256, 0, stream>>>(x, style, weight, g, out);
}
